// Net_66546223284519
// MI455X (gfx1250) — matrix operand order repaired and checked by reading, compiled, not run on hardware
//
#include <hip/hip_runtime.h>
#include <hip/hip_bf16.h>
#include <math.h>

// ---------------------------------------------------------------------------
// Problem constants (fixed by the reference)
// ---------------------------------------------------------------------------
#define NN      60000      // nodes
#define NE      240000     // edges
#define NREL    4
#define NG      128        // graphs
#define DIN     768
#define DH      256
#define NLAYERS 6
#define NSEG    5          // 4 relations + root treated as segment 4
#define NT      8          // N-tiles (16 cols each) per wave -> 16x128 strip

typedef __bf16 v16bf __attribute__((ext_vector_type(16)));
typedef float  v8f   __attribute__((ext_vector_type(8)));

// ---------------------------------------------------------------------------
// Utility: zero a float buffer (grid-stride)
// ---------------------------------------------------------------------------
__global__ void zero_f32(float* __restrict__ p, long long n) {
    long long i = (long long)blockIdx.x * blockDim.x + threadIdx.x;
    long long stride = (long long)gridDim.x * blockDim.x;
    for (; i < n; i += stride) p[i] = 0.0f;
}

// ---------------------------------------------------------------------------
// Edge degree counts per (rel, dst) segment, then inverse
// ---------------------------------------------------------------------------
__global__ void count_edges(const int* __restrict__ dst, const int* __restrict__ rel,
                            float* __restrict__ cnt) {
    int e = blockIdx.x * blockDim.x + threadIdx.x;
    if (e >= NE) return;
    atomicAdd(&cnt[(size_t)rel[e] * NN + dst[e]], 1.0f);
}

__global__ void inv_counts(const float* __restrict__ cnt, float* __restrict__ inv) {
    int i = blockIdx.x * blockDim.x + threadIdx.x;
    if (i >= NREL * NN) return;
    inv[i] = 1.0f / fmaxf(cnt[i], 1.0f);
}

// ---------------------------------------------------------------------------
// Scatter: agg[rel][dst][k] += h[src][k]   (fp32 atomics, float4 gathers)
// One thread handles 4 contiguous features of one edge.
// ---------------------------------------------------------------------------
__global__ void scatter_edges(const int* __restrict__ src, const int* __restrict__ dst,
                              const int* __restrict__ rel, const float* __restrict__ h,
                              float* __restrict__ agg, int K) {
    int kq = K >> 2;                       // float4 chunks per row
    long long idx = (long long)blockIdx.x * blockDim.x + threadIdx.x;
    long long total = (long long)NE * kq;
    if (idx >= total) return;
    int e = (int)(idx / kq);
    int f = (int)(idx % kq) << 2;
    const float4 v = *(const float4*)(h + (size_t)src[e] * K + f);
    float* base = agg + (((size_t)rel[e] * NN + dst[e]) * K + f);
    atomicAdd(base + 0, v.x);
    atomicAdd(base + 1, v.y);
    atomicAdd(base + 2, v.z);
    atomicAdd(base + 3, v.w);
}

// ---------------------------------------------------------------------------
// Pack [5][K][256] fp32 weights (4 relation mats + root) into the bf16
// B-fragment layout (contiguous k order, same as A) for v_wmma_f32_16x16x32_bf16:
//   fragment element j of lane L holds B[k0 + 16*(L>>4) + j][L&15 + n0]
// so we store, for each (seg s, k-block kb, half, col c), the 16 bf16 values
// j=0..15 contiguously -> one 32-byte vector load per lane in the GEMM.
// ---------------------------------------------------------------------------
__global__ void pack_weights(const float* __restrict__ w,     // [4][K][256]
                             const float* __restrict__ root,  // [K][256]
                             __bf16* __restrict__ dstp, int K) {
    int KB = K >> 5;
    int idx = blockIdx.x * blockDim.x + threadIdx.x;
    int total = NSEG * K * DH;
    if (idx >= total) return;
    int c = idx % DH;
    int t = idx / DH;
    int k = t % K;
    int s = t / K;
    float v = (s < NREL) ? w[((size_t)s * K + k) * DH + c]
                         : root[(size_t)k * DH + c];
    int kb = k >> 5, rem = k & 31, half = rem >> 4, j = rem & 15;
    size_t d = ((((size_t)s * KB + kb) * 2 + half) * DH + c) * 16 + j;
    dstp[d] = (__bf16)v;
}

// ---------------------------------------------------------------------------
// RGCN layer GEMM with WMMA (bf16 in, fp32 accumulate):
//   out[n][o] = relu( bias[o] + sum_{s<4} sum_k (agg[s][n][k]*inv[s][n]) * W_s[k][o]
//                              + sum_k h[n][k] * root[k][o] )
// One wave computes a 16x128 output strip: the A fragment (the expensive fp32
// agg read + bf16 convert) is built once per k-step and reused by 8 WMMAs
// against 8 contiguous B fragments.  8 waves / 256-thread block.
// Grid: 3750 M-tiles x 2 half-strips = 7500 waves = 938 blocks.
// ---------------------------------------------------------------------------
__global__ __launch_bounds__(256)
void rgcn_gemm_wmma(const float* __restrict__ agg,   // [4][NN][K] fp32 sums
                    const float* __restrict__ inv,   // [4][NN]    1/max(cnt,1)
                    const float* __restrict__ h,     // [NN][K]    current features
                    const __bf16* __restrict__ wfrag,// packed [5][K/32][2][256][16]
                    const float* __restrict__ bias,  // [256]
                    float* __restrict__ out,         // [NN][256]
                    int K) {
    const int wid = blockIdx.x * 8 + (threadIdx.x >> 5);
    if (wid >= (NN / 16) * (DH / (16 * NT))) return;
    const int mtile = wid >> 1;           // DH/(16*NT) = 2 strips per M-tile
    const int strip = wid & 1;
    const int m0 = mtile * 16;
    const int n0 = strip * (16 * NT);

    const int lane = threadIdx.x & 31;
    const int half = lane >> 4;           // selects K-half of the fragment
    const int l15  = lane & 15;

    const int KB = K >> 5;                // k-blocks of 32
    v8f acc[NT];
#pragma unroll
    for (int t = 0; t < NT; ++t) acc[t] = (v8f){};

    for (int s = 0; s < NSEG; ++s) {
        const int row = m0 + l15;         // A: lane&15 = matrix row for both halves
        const float* arow;
        float scale;
        if (s < NREL) {
            arow  = agg + ((size_t)s * NN + row) * K;
            scale = inv[(size_t)s * NN + row];
        } else {
            arow  = h + (size_t)row * K;
            scale = 1.0f;
        }
        const __bf16* wseg = wfrag + (((size_t)s * KB) * 2) * (DH * 16);

        for (int kb = 0; kb < KB; ++kb) {
            const int k0 = kb * 32;
            // A fragment, contiguous K order (matches packed B):
            //  j=0..15 -> K = k0 + 16*half + j
            const float4* pa0 = (const float4*)(arow + k0 + 16 * half);
            const float4* pa1 = (const float4*)(arow + k0 + 16 * half + 8);
            float4 q0 = pa0[0], q1 = pa0[1];
            float4 q2 = pa1[0], q3 = pa1[1];
            if (kb + 1 < KB)  // hint next A k-block into cache (global_prefetch_b8)
                __builtin_prefetch(arow + k0 + 32 + 8 * half, 0, 1);

            v16bf a;
            a[0]  = (__bf16)(q0.x * scale); a[1]  = (__bf16)(q0.y * scale);
            a[2]  = (__bf16)(q0.z * scale); a[3]  = (__bf16)(q0.w * scale);
            a[4]  = (__bf16)(q1.x * scale); a[5]  = (__bf16)(q1.y * scale);
            a[6]  = (__bf16)(q1.z * scale); a[7]  = (__bf16)(q1.w * scale);
            a[8]  = (__bf16)(q2.x * scale); a[9]  = (__bf16)(q2.y * scale);
            a[10] = (__bf16)(q2.z * scale); a[11] = (__bf16)(q2.w * scale);
            a[12] = (__bf16)(q3.x * scale); a[13] = (__bf16)(q3.y * scale);
            a[14] = (__bf16)(q3.z * scale); a[15] = (__bf16)(q3.w * scale);

            // NT B fragments: each a contiguous 32-byte bf16 vector load,
            // consecutive fragments 512B apart (16 cols x 16 elems x 2B)
            const __bf16* wb = wseg + (((size_t)kb * 2 + half) * DH + (n0 + l15)) * 16;
#pragma unroll
            for (int t = 0; t < NT; ++t) {
                const v16bf b = *(const v16bf*)(wb + (size_t)t * 16 * 16);
                acc[t] = __builtin_amdgcn_wmma_f32_16x16x32_bf16(
                    false, a, false, b, (short)0, acc[t], false, false);
            }
        }
    }

    // D tiles: element v of acc -> row m0 + v + 8*half, col n0 + t*16 + (lane&15)
#pragma unroll
    for (int t = 0; t < NT; ++t) {
        const int col = n0 + t * 16 + l15;
        const float bcol = bias[col];
#pragma unroll
        for (int v = 0; v < 8; ++v) {
            const int row = m0 + v + 8 * half;
            float val = acc[t][v] + bcol;
            out[(size_t)row * DH + col] = val > 0.0f ? val : 0.0f;
        }
    }
}

// ---------------------------------------------------------------------------
// Graph pooling: mean + max over nodes of each graph
// (post-ReLU features are >= 0, so bit-pattern atomicMax with 0-init is exact
//  and also reproduces where(cnt>0, max, 0))
// ---------------------------------------------------------------------------
__global__ void pool_scatter(const float* __restrict__ h, const int* __restrict__ batch,
                             float* __restrict__ psum, int* __restrict__ pmax_bits,
                             float* __restrict__ pcnt) {
    int idx = blockIdx.x * blockDim.x + threadIdx.x;
    if (idx >= NN * DH) return;
    int n = idx >> 8;
    int f = idx & 255;
    int b = batch[n];
    float v = h[(size_t)n * DH + f];
    atomicAdd(&psum[b * DH + f], v);
    atomicMax(&pmax_bits[b * DH + f], __float_as_int(v));
    if (f == 0) atomicAdd(&pcnt[b], 1.0f);
}

__global__ void pool_finalize(const float* __restrict__ psum, const int* __restrict__ pmax_bits,
                              const float* __restrict__ pcnt, float* __restrict__ g) {
    int idx = blockIdx.x * blockDim.x + threadIdx.x;
    if (idx >= NG * DH) return;
    int b = idx >> 8;
    int f = idx & 255;
    float c = fmaxf(pcnt[b], 1.0f);
    g[(size_t)b * (2 * DH) + f]      = psum[idx] / c;
    g[(size_t)b * (2 * DH) + DH + f] = __int_as_float(pmax_bits[idx]);
}

// ---------------------------------------------------------------------------
// Tiny MLP head: [128,512] @ [512,128] +b, relu, @ [128,1] +b, sigmoid
// ---------------------------------------------------------------------------
__global__ void fc1_kernel(const float* __restrict__ g, const float* __restrict__ w,
                           const float* __restrict__ bias, float* __restrict__ t) {
    int idx = blockIdx.x * blockDim.x + threadIdx.x;
    if (idx >= NG * 128) return;
    int gi = idx >> 7;
    int j  = idx & 127;
    const float* gr = g + (size_t)gi * (2 * DH);
    float acc = bias[j];
    for (int i = 0; i < 2 * DH; ++i) acc += gr[i] * w[i * 128 + j];
    t[idx] = fmaxf(acc, 0.0f);
}

__global__ void fc2_kernel(const float* __restrict__ t, const float* __restrict__ w,
                           const float* __restrict__ bias, float* __restrict__ out) {
    int gi = blockIdx.x * blockDim.x + threadIdx.x;
    if (gi >= NG) return;
    float acc = bias[0];
    const float* tr = t + (size_t)gi * 128;
    for (int j = 0; j < 128; ++j) acc += tr[j] * w[j];
    out[gi] = 1.0f / (1.0f + expf(-acc));
}

// ---------------------------------------------------------------------------
// Host-side orchestration
// ---------------------------------------------------------------------------
static inline size_t alignup(size_t x) { return (x + 255) & ~(size_t)255; }

extern "C" void kernel_launch(void* const* d_in, const int* in_sizes, int n_in,
                              void* d_out, int out_size, void* d_ws, size_t ws_size,
                              hipStream_t stream) {
    (void)in_sizes; (void)n_in; (void)out_size; (void)ws_size;

    const float* x        = (const float*)d_in[0];
    const int*   eidx     = (const int*)d_in[1];     // [2][NE]: src, dst
    const int*   eattr    = (const int*)d_in[2];     // [NE]
    const int*   batch    = (const int*)d_in[3];     // [NN]
    const float* w0       = (const float*)d_in[4];   // [4][768][256]
    const float* root0    = (const float*)d_in[5];   // [768][256]
    const float* b0       = (const float*)d_in[6];   // [256]
    const float* w_rest   = (const float*)d_in[7];   // [5][4][256][256]
    const float* root_rest= (const float*)d_in[8];   // [5][256][256]
    const float* b_rest   = (const float*)d_in[9];   // [5][256]
    const float* fc1_w    = (const float*)d_in[10];  // [512][128]
    const float* fc1_b    = (const float*)d_in[11];  // [128]
    const float* fc2_w    = (const float*)d_in[12];  // [128][1]
    const float* fc2_b    = (const float*)d_in[13];  // [1]
    float*       out      = (float*)d_out;           // [128]

    const int* esrc = eidx;
    const int* edst = eidx + NE;

    // ---- carve workspace -------------------------------------------------
    char* ws = (char*)d_ws;
    size_t off = 0;
    auto carve = [&](size_t bytes) { char* p = ws + off; off += alignup(bytes); return p; };

    float*  agg   = (float*)carve((size_t)NREL * NN * DIN * sizeof(float)); // reused all layers
    float*  cnt   = (float*)carve((size_t)NREL * NN * sizeof(float));
    float*  invc  = (float*)carve((size_t)NREL * NN * sizeof(float));
    float*  hA    = (float*)carve((size_t)NN * DH * sizeof(float));
    float*  hB    = (float*)carve((size_t)NN * DH * sizeof(float));
    __bf16* wf0   = (__bf16*)carve((size_t)NSEG * DIN * DH * sizeof(__bf16));
    __bf16* wfR   = (__bf16*)carve((size_t)(NLAYERS - 1) * NSEG * DH * DH * sizeof(__bf16));
    float*  psum  = (float*)carve((size_t)NG * DH * sizeof(float));
    int*    pmaxb = (int*)carve((size_t)NG * DH * sizeof(int));
    float*  pcnt  = (float*)carve((size_t)NG * sizeof(float));
    float*  gfeat = (float*)carve((size_t)NG * 2 * DH * sizeof(float));
    float*  t1    = (float*)carve((size_t)NG * 128 * sizeof(float));

    // ---- pack weights into bf16 WMMA-fragment layout ---------------------
    {
        int total0 = NSEG * DIN * DH;
        pack_weights<<<(total0 + 255) / 256, 256, 0, stream>>>(w0, root0, wf0, DIN);
        int totalR = NSEG * DH * DH;
        for (int i = 0; i < NLAYERS - 1; ++i) {
            pack_weights<<<(totalR + 255) / 256, 256, 0, stream>>>(
                w_rest + (size_t)i * NREL * DH * DH,
                root_rest + (size_t)i * DH * DH,
                wfR + (size_t)i * NSEG * DH * DH, DH);
        }
    }

    // ---- per-(rel,dst) degree counts (constant across layers) ------------
    zero_f32<<<2048, 256, 0, stream>>>(cnt, (long long)NREL * NN);
    count_edges<<<(NE + 255) / 256, 256, 0, stream>>>(edst, eattr, cnt);
    inv_counts<<<(NREL * NN + 255) / 256, 256, 0, stream>>>(cnt, invc);

    // ---- 6 RGCN layers ---------------------------------------------------
    const float* hin = x;
    float* hout = hA;
    for (int layer = 0; layer < NLAYERS; ++layer) {
        const int K = (layer == 0) ? DIN : DH;
        const __bf16* wf = (layer == 0) ? wf0 : (wfR + (size_t)(layer - 1) * NSEG * DH * DH);
        const float* bias = (layer == 0) ? b0 : (b_rest + (size_t)(layer - 1) * DH);

        long long aggN = (long long)NREL * NN * K;
        zero_f32<<<8192, 256, 0, stream>>>(agg, aggN);

        long long sthreads = (long long)NE * (K >> 2);
        scatter_edges<<<(unsigned)((sthreads + 255) / 256), 256, 0, stream>>>(
            esrc, edst, eattr, hin, agg, K);

        // 3750 M-tiles x 2 half-strips = 7500 waves, 8 waves per block
        int nwaves = (NN / 16) * (DH / (16 * NT));
        rgcn_gemm_wmma<<<(nwaves + 7) / 8, 256, 0, stream>>>(
            agg, invc, hin, wf, bias, hout, K);

        hin  = hout;
        hout = (hout == hA) ? hB : hA;
    }

    // ---- pooling ---------------------------------------------------------
    zero_f32<<<256, 256, 0, stream>>>(psum, (long long)NG * DH);
    zero_f32<<<256, 256, 0, stream>>>((float*)pmaxb, (long long)NG * DH);
    zero_f32<<<1, 256, 0, stream>>>(pcnt, (long long)NG);
    pool_scatter<<<(NN * DH + 255) / 256, 256, 0, stream>>>(hin, batch, psum, pmaxb, pcnt);
    pool_finalize<<<(NG * DH + 255) / 256, 256, 0, stream>>>(psum, pmaxb, pcnt, gfeat);

    // ---- MLP head --------------------------------------------------------
    fc1_kernel<<<(NG * 128 + 255) / 256, 256, 0, stream>>>(gfeat, fc1_w, fc1_b, t1);
    fc2_kernel<<<1, NG, 0, stream>>>(t1, fc2_w, fc2_b, out);
}
